// ConvEncodeDecodeLargeVQVAE_86406152061158
// MI455X (gfx1250) — matrix operand order repaired and checked by reading, not compiled, not run on hardware
//
#include <hip/hip_runtime.h>
#include <hip/hip_bf16.h>
#include <stdint.h>
#include <stddef.h>

// ---------------------------------------------------------------------------
// CDNA5 / gfx1250 VQ-VAE forward.  wave32, WMMA 16x16x32 f16 for all
// GEMM-shaped work (fc21, fc22, fc3, VQ codebook search); direct VALU
// kernels with folded BN+ReLU for the small spatial (de)convolutions.
// ---------------------------------------------------------------------------

typedef _Float16 v8h  __attribute__((ext_vector_type(8)));
typedef _Float16 v16h __attribute__((ext_vector_type(16)));
typedef float    v8f  __attribute__((ext_vector_type(8)));

#define BN_EPS 1e-5f

__device__ __forceinline__ float bn_apply(float x, const float* g, const float* b,
                                          const float* m, const float* v, int c) {
  float s = g[c] * rsqrtf(v[c] + BN_EPS);
  return x * s + (b[c] - m[c] * s);
}

// ---------------------------------------------------------------------------
// Generic direct Conv2d (NCHW) + BN + ReLU, one thread per output element.
// ---------------------------------------------------------------------------
__global__ void conv_bn_relu_kernel(const float* __restrict__ in, const float* __restrict__ w,
    const float* __restrict__ bias, const float* __restrict__ bg, const float* __restrict__ bb,
    const float* __restrict__ bm, const float* __restrict__ bv, float* __restrict__ out,
    int N, int Cin, int Hin, int Win, int Cout, int Hout, int Wout, int Kk, int stride, int pad)
{
  int idx = blockIdx.x * blockDim.x + threadIdx.x;
  int total = N * Cout * Hout * Wout;
  if (idx >= total) return;
  int ow_ = idx % Wout; int t = idx / Wout;
  int oh  = t % Hout;   t /= Hout;
  int oc  = t % Cout;   int n = t / Cout;
  float acc = bias[oc];
  for (int ic = 0; ic < Cin; ++ic) {
    const float* ip = in + (size_t)(n * Cin + ic) * Hin * Win;
    const float* wp = w  + (size_t)(oc * Cin + ic) * Kk * Kk;
    for (int kh = 0; kh < Kk; ++kh) {
      int ih = oh * stride - pad + kh;
      if (ih < 0 || ih >= Hin) continue;
      for (int kw = 0; kw < Kk; ++kw) {
        int iw = ow_ * stride - pad + kw;
        if (iw < 0 || iw >= Win) continue;
        acc += ip[ih * Win + iw] * wp[kh * Kk + kw];
      }
    }
  }
  acc = bn_apply(acc, bg, bb, bm, bv, oc);
  out[idx] = fmaxf(acc, 0.f);
}

// ---------------------------------------------------------------------------
// 1x1 Conv (16->128) + BN + ReLU, writing f16 (this IS hl = h.reshape(N,-1)).
// ---------------------------------------------------------------------------
__global__ void conv1x1_bn_relu_f16_kernel(const float* __restrict__ in, const float* __restrict__ w,
    const float* __restrict__ bias, const float* __restrict__ bg, const float* __restrict__ bb,
    const float* __restrict__ bm, const float* __restrict__ bv, _Float16* __restrict__ out16,
    int N, int Cin, int Cout, int HW)
{
  int idx = blockIdx.x * blockDim.x + threadIdx.x;
  int total = N * Cout * HW;
  if (idx >= total) return;
  int hw = idx % HW; int t = idx / HW;
  int oc = t % Cout; int n = t / Cout;
  float acc = bias[oc];
  for (int ic = 0; ic < Cin; ++ic)
    acc += in[(size_t)(n * Cin + ic) * HW + hw] * w[oc * Cin + ic];
  acc = bn_apply(acc, bg, bb, bm, bv, oc);
  out16[idx] = (_Float16)fmaxf(acc, 0.f);
}

// ---------------------------------------------------------------------------
// 1x1 ConvTranspose (weight [Cin,Cout]) + BN + ReLU.
// ---------------------------------------------------------------------------
__global__ void convT1x1_bn_relu_kernel(const float* __restrict__ in, const float* __restrict__ w,
    const float* __restrict__ bias, const float* __restrict__ bg, const float* __restrict__ bb,
    const float* __restrict__ bm, const float* __restrict__ bv, float* __restrict__ out,
    int N, int Cin, int Cout, int HW)
{
  int idx = blockIdx.x * blockDim.x + threadIdx.x;
  int total = N * Cout * HW;
  if (idx >= total) return;
  int hw = idx % HW; int t = idx / HW;
  int oc = t % Cout; int n = t / Cout;
  float acc = bias[oc];
  for (int ic = 0; ic < Cin; ++ic)
    acc += in[(size_t)(n * Cin + ic) * HW + hw] * w[ic * Cout + oc];
  acc = bn_apply(acc, bg, bb, bm, bv, oc);
  out[idx] = fmaxf(acc, 0.f);
}

// ---------------------------------------------------------------------------
// cw2: 1x1 ConvTranspose 32->32, no BN.  Writes z_e_x (NCHW f32) and the
// pixel-major f16 transpose zt16[p][c] used by the WMMA codebook search.
// ---------------------------------------------------------------------------
__global__ void convT1x1_dual_kernel(const float* __restrict__ in, const float* __restrict__ w,
    const float* __restrict__ bias, float* __restrict__ zex, _Float16* __restrict__ zt16,
    int N, int Cin, int Cout, int HW)
{
  int idx = blockIdx.x * blockDim.x + threadIdx.x;
  int total = N * Cout * HW;
  if (idx >= total) return;
  int hw = idx % HW; int t = idx / HW;
  int oc = t % Cout; int n = t / Cout;
  float acc = bias[oc];
  for (int ic = 0; ic < Cin; ++ic)
    acc += in[(size_t)(n * Cin + ic) * HW + hw] * w[ic * Cout + oc];
  zex[idx] = acc;
  zt16[(size_t)(n * HW + hw) * Cout + oc] = (_Float16)acc;
}

// ---------------------------------------------------------------------------
// Generic ConvTranspose2d (weight [Cin,Cout,K,K]) + BN + ReLU.
// ---------------------------------------------------------------------------
__global__ void convT_bn_relu_kernel(const float* __restrict__ in, const float* __restrict__ w,
    const float* __restrict__ bias, const float* __restrict__ bg, const float* __restrict__ bb,
    const float* __restrict__ bm, const float* __restrict__ bv, float* __restrict__ out,
    int N, int Cin, int Hin, int Win, int Cout, int Hout, int Wout, int Kk, int stride, int pad)
{
  int idx = blockIdx.x * blockDim.x + threadIdx.x;
  int total = N * Cout * Hout * Wout;
  if (idx >= total) return;
  int ow_ = idx % Wout; int t = idx / Wout;
  int oh  = t % Hout;   t /= Hout;
  int oc  = t % Cout;   int n = t / Cout;
  float acc = bias[oc];
  for (int ic = 0; ic < Cin; ++ic) {
    const float* ip = in + (size_t)(n * Cin + ic) * Hin * Win;
    const float* wp = w  + (size_t)(ic * Cout + oc) * Kk * Kk;
    for (int kh = 0; kh < Kk; ++kh) {
      int iht = oh + pad - kh;
      if (iht < 0 || (iht % stride)) continue;
      int ih = iht / stride;
      if (ih >= Hin) continue;
      for (int kw = 0; kw < Kk; ++kw) {
        int iwt = ow_ + pad - kw;
        if (iwt < 0 || (iwt % stride)) continue;
        int iw = iwt / stride;
        if (iw >= Win) continue;
        acc += ip[ih * Win + iw] * wp[kh * Kk + kw];
      }
    }
  }
  acc = bn_apply(acc, bg, bb, bm, bv, oc);
  out[idx] = fmaxf(acc, 0.f);
}

// ---------------------------------------------------------------------------
// Fused decoder tail: dw4 (convT k2 s2 p0: exactly one tap per output pixel)
// + BN4 + ReLU + ow (1x1 convT 16->1).  Avoids the 152 MB g4 tensor.
// ---------------------------------------------------------------------------
__global__ void decode_final_kernel(const float* __restrict__ g3, const float* __restrict__ w4,
    const float* __restrict__ b4, const float* __restrict__ bg, const float* __restrict__ bb,
    const float* __restrict__ bm, const float* __restrict__ bv,
    const float* __restrict__ oww, const float* __restrict__ obb,
    float* __restrict__ xt, int N)
{
  int idx = blockIdx.x * blockDim.x + threadIdx.x;
  int total = N * 68 * 68;
  if (idx >= total) return;
  int ow_ = idx % 68; int t = idx / 68;
  int oh  = t % 68;   int n = t / 68;
  int kh = oh & 1, kw = ow_ & 1;
  int ih = oh >> 1, iw = ow_ >> 1;          // always in [0,34)
  float x = obb[0];
  for (int c = 0; c < 16; ++c) {
    float acc = b4[c];
    for (int i = 0; i < 16; ++i)
      acc += g3[(((size_t)n * 16 + i) * 34 + ih) * 34 + iw] * w4[((i * 16 + c) * 2 + kh) * 2 + kw];
    acc = bn_apply(acc, bg, bb, bm, bv, c);
    x += oww[c] * fmaxf(acc, 0.f);
  }
  xt[idx] = x;
}

// ---------------------------------------------------------------------------
// Transpose+cast: in [K,N] f32  ->  out [N,K] f16 (so B-tile rows are
// K-contiguous and can be fetched with b128 loads per lane).
// ---------------------------------------------------------------------------
__global__ void castT_kernel(const float* __restrict__ in, _Float16* __restrict__ out, int K, int N)
{
  int idx = blockIdx.x * blockDim.x + threadIdx.x;
  if (idx >= K * N) return;
  int k = idx / N, n = idx % N;
  out[(size_t)n * K + k] = (_Float16)in[idx];
}

// emb -> f16 copy + per-code squared norm e2[k] = ||emb[k]||^2
__global__ void emb_prep_kernel(const float* __restrict__ emb, _Float16* __restrict__ emb16,
                                float* __restrict__ e2, int Kcodes, int C)
{
  int k = blockIdx.x * blockDim.x + threadIdx.x;
  if (k >= Kcodes) return;
  float s = 0.f;
  for (int c = 0; c < C; ++c) {
    float v = emb[(size_t)k * C + c];
    emb16[(size_t)k * C + c] = (_Float16)v;
    s += v * v;
  }
  e2[k] = s;
}

// z_q gather: z_q_x[n,c,hw] = emb[lat[n,hw], c]
__global__ void zq_gather_kernel(const int* __restrict__ lat, const float* __restrict__ emb,
                                 float* __restrict__ zqx, int N, int C, int HW)
{
  int idx = blockIdx.x * blockDim.x + threadIdx.x;
  int total = N * C * HW;
  if (idx >= total) return;
  int hw = idx % HW; int t = idx / HW;
  int c  = t % C;    int n = t / C;
  zqx[idx] = emb[(size_t)lat[n * HW + hw] * C + c];
}

// ---------------------------------------------------------------------------
// WMMA f16 GEMM:  out[M,N] = A[M,K] * Bt[N,K]^T + bias, optional ReLU,
// optional duplicate f32 output (z==mu) and f16 output (feeds next GEMM).
// One 16x16 output tile per wave; K in steps of 32 (one v_wmma each).
// A/B lane tiles per ISA layout:
//   A: lane(0-15)=row M, halves hold K {h*8..h*8+7, 16+h*8..};  two b128 loads
//   B: lane(0-15)=col N, same K order as A;                      two b128 loads
// ---------------------------------------------------------------------------
__global__ void gemm_wmma_kernel(const _Float16* __restrict__ A, const _Float16* __restrict__ Bt,
    const float* __restrict__ bias, float* __restrict__ out, float* __restrict__ out2,
    _Float16* __restrict__ out16, int M, int N, int K, int relu)
{
  int wave = blockIdx.x * (blockDim.x >> 5) + (threadIdx.x >> 5);
  int tilesN = N >> 4;
  int numTiles = (M >> 4) * tilesN;
  if (wave >= numTiles) return;                 // wave-uniform: EXEC stays full
  int tm = wave / tilesN, tn = wave % tilesN;
  int lane = threadIdx.x & 31;
  int r = lane & 15, half = lane >> 4;

  const _Float16* Arow = A  + (size_t)(tm * 16 + r) * K;
  const _Float16* Brow = Bt + (size_t)(tn * 16 + r) * K;

  v8f acc = {};
  for (int kk = 0; kk < K; kk += 32) {
    v8h alo = *(const v8h*)(Arow + kk + half * 8);
    v8h ahi = *(const v8h*)(Arow + kk + 16 + half * 8);
    v8h b0  = *(const v8h*)(Brow + kk + half * 8);
    v8h b1  = *(const v8h*)(Brow + kk + 16 + half * 8);
    v16h a, b;
#pragma unroll
    for (int j = 0; j < 8; ++j) { a[j] = alo[j]; a[j + 8] = ahi[j]; b[j] = b0[j]; b[j + 8] = b1[j]; }
    acc = __builtin_amdgcn_wmma_f32_16x16x32_f16(false, a, false, b, (short)0, acc, false, false);
  }

  int col = tn * 16 + r;
  float bvv = bias ? bias[col] : 0.f;
#pragma unroll
  for (int q = 0; q < 8; ++q) {
    int row = tm * 16 + half * 8 + q;           // D layout: lanes 0-15 -> M=q, 16-31 -> M=q+8
    float v = acc[q] + bvv;
    if (relu) v = fmaxf(v, 0.f);
    size_t o = (size_t)row * N + col;
    out[o] = v;
    if (out2)  out2[o]  = v;
    if (out16) out16[o] = (_Float16)v;
  }
}

// ---------------------------------------------------------------------------
// VQ codebook search: one wave owns 16 pixels (rows).  K=32 => a single
// v_wmma per 16-code tile; loop over 32 code tiles keeping running
// argmin of (e2[k] - 2*dot) per row; cross-lane shfl_xor reduction with
// first-index tie-break (matches jnp.argmin).
// ---------------------------------------------------------------------------
__global__ void vq_argmin_kernel(const _Float16* __restrict__ zt, const _Float16* __restrict__ emb16,
    const float* __restrict__ e2, int* __restrict__ lat, float* __restrict__ latf, int P)
{
  int wave = blockIdx.x * (blockDim.x >> 5) + (threadIdx.x >> 5);
  if (wave >= (P >> 4)) return;
  int lane = threadIdx.x & 31;
  int r = lane & 15, half = lane >> 4;

  const _Float16* Arow = zt + (size_t)(wave * 16 + r) * 32;
  v16h a;
  {
    v8h alo = *(const v8h*)(Arow + half * 8);
    v8h ahi = *(const v8h*)(Arow + 16 + half * 8);
#pragma unroll
    for (int j = 0; j < 8; ++j) { a[j] = alo[j]; a[j + 8] = ahi[j]; }
  }

  float best[8]; int bidx[8];
#pragma unroll
  for (int q = 0; q < 8; ++q) { best[q] = 3.4e38f; bidx[q] = 0; }

  for (int nt = 0; nt < 32; ++nt) {
    const _Float16* Brow = emb16 + (size_t)(nt * 16 + r) * 32;
    v8h b0 = *(const v8h*)(Brow + half * 8);
    v8h b1 = *(const v8h*)(Brow + 16 + half * 8);
    v16h b;
#pragma unroll
    for (int j = 0; j < 8; ++j) { b[j] = b0[j]; b[j + 8] = b1[j]; }
    v8f c = {};
    c = __builtin_amdgcn_wmma_f32_16x16x32_f16(false, a, false, b, (short)0, c, false, false);
    int code = nt * 16 + r;
    float ev = e2[code];
#pragma unroll
    for (int q = 0; q < 8; ++q) {
      float s = ev - 2.f * c[q];                // ||z||^2 term constant per row: drop
      if (s < best[q]) { best[q] = s; bidx[q] = code; }
    }
  }
  // reduce across the 16 lanes that hold different code columns (same half)
#pragma unroll
  for (int off = 1; off < 16; off <<= 1) {
#pragma unroll
    for (int q = 0; q < 8; ++q) {
      float ob = __shfl_xor(best[q], off, 32);
      int   oi = __shfl_xor(bidx[q], off, 32);
      if (ob < best[q] || (ob == best[q] && oi < bidx[q])) { best[q] = ob; bidx[q] = oi; }
    }
  }
  if (r == 0) {
#pragma unroll
    for (int q = 0; q < 8; ++q) {
      int row = wave * 16 + half * 8 + q;
      lat[row]  = bidx[q];
      latf[row] = (float)bidx[q];
    }
  }
}

// ---------------------------------------------------------------------------
// Host launcher
// ---------------------------------------------------------------------------
extern "C" void kernel_launch(void* const* d_in, const int* in_sizes, int n_in,
                              void* d_out, int out_size, void* d_ws, size_t ws_size,
                              hipStream_t stream) {
  (void)in_sizes; (void)n_in; (void)out_size; (void)ws_size;
  const float* x     = (const float*)d_in[0];
  const float* ew1   = (const float*)d_in[1];  const float* eb1 = (const float*)d_in[2];
  const float* ew2   = (const float*)d_in[3];  const float* eb2 = (const float*)d_in[4];
  const float* ew3   = (const float*)d_in[5];  const float* eb3 = (const float*)d_in[6];
  const float* ew4   = (const float*)d_in[7];  const float* eb4 = (const float*)d_in[8];
  const float* fc21w = (const float*)d_in[9];  const float* fc21b = (const float*)d_in[10];
  const float* fc22w = (const float*)d_in[11]; const float* fc22b = (const float*)d_in[12];
  const float* fc3w  = (const float*)d_in[13]; const float* fc3b  = (const float*)d_in[14];
  const float* cw1   = (const float*)d_in[15]; const float* cb1 = (const float*)d_in[16];
  const float* cw2   = (const float*)d_in[17]; const float* cb2 = (const float*)d_in[18];
  const float* dw1   = (const float*)d_in[19]; const float* db1 = (const float*)d_in[20];
  const float* dw2   = (const float*)d_in[21]; const float* db2 = (const float*)d_in[22];
  const float* dw3   = (const float*)d_in[23]; const float* db3 = (const float*)d_in[24];
  const float* dw4   = (const float*)d_in[25]; const float* db4 = (const float*)d_in[26];
  const float* oww   = (const float*)d_in[27]; const float* obb = (const float*)d_in[28];
  const float* bn1[4]  = {(const float*)d_in[29], (const float*)d_in[30], (const float*)d_in[31], (const float*)d_in[32]};
  const float* bn2[4]  = {(const float*)d_in[33], (const float*)d_in[34], (const float*)d_in[35], (const float*)d_in[36]};
  const float* bn3[4]  = {(const float*)d_in[37], (const float*)d_in[38], (const float*)d_in[39], (const float*)d_in[40]};
  const float* bn4[4]  = {(const float*)d_in[41], (const float*)d_in[42], (const float*)d_in[43], (const float*)d_in[44]};
  const float* cbn1[4] = {(const float*)d_in[45], (const float*)d_in[46], (const float*)d_in[47], (const float*)d_in[48]};
  const float* dbn1[4] = {(const float*)d_in[49], (const float*)d_in[50], (const float*)d_in[51], (const float*)d_in[52]};
  const float* dbn2[4] = {(const float*)d_in[53], (const float*)d_in[54], (const float*)d_in[55], (const float*)d_in[56]};
  const float* dbn3[4] = {(const float*)d_in[57], (const float*)d_in[58], (const float*)d_in[59], (const float*)d_in[60]};
  const float* dbn4[4] = {(const float*)d_in[61], (const float*)d_in[62], (const float*)d_in[63], (const float*)d_in[64]};
  const float* emb   = (const float*)d_in[65];

  const int N = 512, HW9 = 81, P = N * HW9;      // 41472 latent pixels

  // ---- output layout (flat concat, reference return order) ----
  float* o_xt     = (float*)d_out;                         // [512,1,68,68]
  float* o_z      = o_xt + (size_t)N * 68 * 68;            // [512,64]
  float* o_mu     = o_z  + (size_t)N * 64;
  float* o_logvar = o_mu + (size_t)N * 64;
  float* o_zex    = o_logvar + (size_t)N * 64;             // [512,32,9,9]
  float* o_zqx    = o_zex + (size_t)N * 32 * HW9;
  float* o_lat    = o_zqx + (size_t)N * 32 * HW9;          // [512,9,9] (as float)

  // ---- workspace layout (slot0/1/2 reused between encoder & decoder) ----
  char* ws = (char*)d_ws;
  size_t off = 0;
  auto alloc = [&](size_t bytes) -> char* {
    char* p = ws + off; off = (off + bytes + 255) & ~(size_t)255; return p;
  };
  float*    s0    = (float*)alloc((size_t)N * 16 * 34 * 34 * 4);   // h1 [16,32,32] then g3 [16,34,34]
  float*    s1    = (float*)alloc((size_t)N * 16 * 18 * 18 * 4);   // h2 [16,16,16] then g2 [16,18,18]
  float*    s2    = (float*)alloc((size_t)N * 16 * HW9 * 4);       // h3 [16,9,9]  then g1 [16,9,9]
  _Float16* hl16  = (_Float16*)alloc((size_t)N * 10368 * 2);
  _Float16* w21t  = (_Float16*)alloc((size_t)64 * 10368 * 2);
  _Float16* w22t  = (_Float16*)alloc((size_t)64 * 10368 * 2);
  _Float16* w3t   = (_Float16*)alloc((size_t)10368 * 64 * 2);
  _Float16* z16   = (_Float16*)alloc((size_t)N * 64 * 2);
  float*    co    = (float*)alloc((size_t)N * 10368 * 4);
  float*    ze    = (float*)alloc((size_t)N * 32 * HW9 * 4);
  _Float16* zt16  = (_Float16*)alloc((size_t)P * 32 * 2);
  _Float16* emb16 = (_Float16*)alloc((size_t)512 * 32 * 2);
  float*    e2    = (float*)alloc((size_t)512 * 4);
  int*      lat   = (int*)alloc((size_t)P * 4);

  const int BT = 256;
  auto grid = [](long long n) { return (unsigned)((n + 255) / 256); };

  // ---- weight prep (independent of activations) ----
  castT_kernel<<<grid(10368LL * 64), BT, 0, stream>>>(fc21w, w21t, 10368, 64);
  castT_kernel<<<grid(10368LL * 64), BT, 0, stream>>>(fc22w, w22t, 10368, 64);
  castT_kernel<<<grid(64LL * 10368), BT, 0, stream>>>(fc3w,  w3t,  64, 10368);
  emb_prep_kernel<<<grid(512), BT, 0, stream>>>(emb, emb16, e2, 512, 32);

  // ---- encoder ----
  conv_bn_relu_kernel<<<grid((long long)N * 16 * 32 * 32), BT, 0, stream>>>(
      x, ew1, eb1, bn1[0], bn1[1], bn1[2], bn1[3], s0, N, 1, 64, 64, 16, 32, 32, 4, 2, 1);
  conv_bn_relu_kernel<<<grid((long long)N * 16 * 16 * 16), BT, 0, stream>>>(
      s0, ew2, eb2, bn2[0], bn2[1], bn2[2], bn2[3], s1, N, 16, 32, 32, 16, 16, 16, 4, 2, 1);
  conv_bn_relu_kernel<<<grid((long long)N * 16 * HW9), BT, 0, stream>>>(
      s1, ew3, eb3, bn3[0], bn3[1], bn3[2], bn3[3], s2, N, 16, 16, 16, 16, 9, 9, 2, 2, 1);
  conv1x1_bn_relu_f16_kernel<<<grid((long long)N * 128 * HW9), BT, 0, stream>>>(
      s2, ew4, eb4, bn4[0], bn4[1], bn4[2], bn4[3], hl16, N, 16, 128, HW9);

  // ---- FC head (WMMA): mu (=z), logvar ----
  {
    int tiles = (N / 16) * (64 / 16);                    // 128 wave tiles
    gemm_wmma_kernel<<<(tiles + 7) / 8, 256, 0, stream>>>(
        hl16, w21t, fc21b, o_mu, o_z, z16, N, 64, 10368, 0);
    gemm_wmma_kernel<<<(tiles + 7) / 8, 256, 0, stream>>>(
        hl16, w22t, fc22b, o_logvar, nullptr, nullptr, N, 64, 10368, 0);
  }
  // ---- fc3 (WMMA) + ReLU -> co [512,10368] ----
  {
    int tiles = (N / 16) * (10368 / 16);                 // 20736 wave tiles
    gemm_wmma_kernel<<<(tiles + 7) / 8, 256, 0, stream>>>(
        z16, w3t, fc3b, co, nullptr, nullptr, N, 10368, 64, 1);
  }

  // ---- vq_decode ----
  convT1x1_bn_relu_kernel<<<grid((long long)N * 32 * HW9), BT, 0, stream>>>(
      co, cw1, cb1, cbn1[0], cbn1[1], cbn1[2], cbn1[3], ze, N, 128, 32, HW9);
  convT1x1_dual_kernel<<<grid((long long)N * 32 * HW9), BT, 0, stream>>>(
      ze, cw2, cb2, o_zex, zt16, N, 32, 32, HW9);

  // ---- codebook search (WMMA) + gather ----
  {
    int tiles = P / 16;                                  // 2592 wave tiles
    vq_argmin_kernel<<<(tiles + 7) / 8, 256, 0, stream>>>(zt16, emb16, e2, lat, o_lat, P);
  }
  zq_gather_kernel<<<grid((long long)N * 32 * HW9), BT, 0, stream>>>(lat, emb, o_zqx, N, 32, HW9);

  // ---- decoder ----
  convT1x1_bn_relu_kernel<<<grid((long long)N * 16 * HW9), BT, 0, stream>>>(
      o_zqx, dw1, db1, dbn1[0], dbn1[1], dbn1[2], dbn1[3], s2, N, 32, 16, HW9);
  convT_bn_relu_kernel<<<grid((long long)N * 16 * 18 * 18), BT, 0, stream>>>(
      s2, dw2, db2, dbn2[0], dbn2[1], dbn2[2], dbn2[3], s1, N, 16, 9, 9, 16, 18, 18, 4, 2, 1);
  convT_bn_relu_kernel<<<grid((long long)N * 16 * 34 * 34), BT, 0, stream>>>(
      s1, dw3, db3, dbn3[0], dbn3[1], dbn3[2], dbn3[3], s0, N, 16, 18, 18, 16, 34, 34, 2, 2, 1);
  decode_final_kernel<<<grid((long long)N * 68 * 68), BT, 0, stream>>>(
      s0, dw4, db4, dbn4[0], dbn4[1], dbn4[2], dbn4[3], oww, obb, o_xt, N);
}
